// ProcessingBlock_4861902979744
// MI455X (gfx1250) — matrix operand order repaired and checked by reading, not compiled, not run on hardware
//
#include <hip/hip_runtime.h>

typedef __attribute__((ext_vector_type(16))) _Float16 v16h;
typedef __attribute__((ext_vector_type(8)))  _Float16 v8h;
typedef __attribute__((ext_vector_type(8)))  float    v8f;

#define NB   32
#define PP   2048
#define KNN  8
#define NPTS (NB*PP)          // 65536
#define BNRS 0.99950037469f   // 1/sqrt(1+1e-3)

// workspace layout (bytes)
//   [0, 40960)                 : swizzled f16 B tiles (40 tiles * 512 f16)
//   [131072, 131072+2MB)       : idx int32 [NPTS][8]
//   [131072+2MB, +6MB)         : lift f32 [NPTS][24]
#define WS_IDX_OFF  131072
#define WS_LIFT_OFF (131072 + (size_t)NPTS*KNN*4)

// B-tile table: a1_w2: tiles 0..1 | bx_w: 2..5 | bd1_pw: 6..13 | bd2_pw: 14..21 | c_pw: 22..39

// branch-free ELU: max(x,0) + (exp(min(x,0)) - 1)
__device__ __forceinline__ float eluf(float x) {
    return fmaxf(x, 0.f) + (__expf(fminf(x, 0.f)) - 1.f);
}

__device__ __forceinline__ v8f wmma_f16(v16h a, v16h b, v8f c) {
    return __builtin_amdgcn_wmma_f32_16x16x32_f16(false, a, false, b, (short)0, c, false, false);
}

// A tile: 16x32 f16, lane L row = L&15; elems 0..7 -> K=(L>>4)*8 .. +7 ; elems 8..15 -> K=16+(L>>4)*8 .. +7
__device__ __forceinline__ v16h load_a(const _Float16* base, int stride, int row, int kofs, int lane) {
    int h = lane >> 4;
    const _Float16* p = base + row * stride + kofs + h * 8;
    v8h lo = *(const v8h*)p;
    v8h hi = *(const v8h*)(p + 16);
    return __builtin_shufflevector(lo, hi, 0,1,2,3,4,5,6,7,8,9,10,11,12,13,14,15);
}

// B tile: pre-swizzled so lane L holds 16 f16 of column (L&15): elems 0..7 -> K=(L>>4)*8..+7 ; elems 8..15 -> K=16+(L>>4)*8..+7
__device__ __forceinline__ v16h load_b(const _Float16* swz, int tile, int lane) {
    return *(const v16h*)(swz + tile * 512 + lane * 16);
}

// -------------------- kernel 1: swizzle weights into WMMA B-tile layout (f16) --------------------
__global__ void prep_weights(const float* __restrict__ a1w2, const float* __restrict__ bxw,
                             const float* __restrict__ pw1, const float* __restrict__ pw2,
                             const float* __restrict__ cpw, _Float16* __restrict__ swz) {
    int e = blockIdx.x * blockDim.x + threadIdx.x;     // 0 .. 40*512-1
    if (e >= 40 * 512) return;
    int t = e >> 9, r = e & 511;
    int lanev = r >> 4, s = r & 15;
    int nl = lanev & 15;
    int k = ((lanev >> 4) << 3) + (s & 7) + ((s >> 3) << 4); // 0..31 within K-subtile
    float v = 0.f;
    if (t < 2) {                                       // a1_w2 [24][24], pad K/N to 32
        int nn = t * 16 + nl;
        if (k < 24 && nn < 24) v = a1w2[k * 24 + nn];
    } else if (t < 6) {                                // bx_w [24][64] (k*3+c rows), pad K to 32
        int nn = (t - 2) * 16 + nl;
        if (k < 24) v = bxw[k * 64 + nn];
    } else if (t < 14) {                               // bd1_pw [64][64]
        int tt = t - 6; int kt = tt >> 2, nt = tt & 3;
        v = pw1[(kt * 32 + k) * 64 + nt * 16 + nl];
    } else if (t < 22) {                               // bd2_pw [64][64]
        int tt = t - 14; int kt = tt >> 2, nt = tt & 3;
        v = pw2[(kt * 32 + k) * 64 + nt * 16 + nl];
    } else {                                           // c_pw [192][48]
        int tt = t - 22; int kt = tt / 3, nt = tt % 3;
        v = cpw[(kt * 32 + k) * 48 + nt * 16 + nl];
    }
    swz[e] = (_Float16)v;
}

// -------------------- kernel 2: fts_lift = elu(fts @ dense0_w) --------------------
__global__ void lift_kernel(const float* __restrict__ fts, const float* __restrict__ w,
                            float* __restrict__ lift) {
    int g = blockIdx.x * blockDim.x + threadIdx.x;
    if (g >= NPTS) return;
    float x = fts[g * 3 + 0], y = fts[g * 3 + 1], z = fts[g * 3 + 2];
    #pragma unroll
    for (int o = 0; o < 24; o++) {
        float v = x * w[o] + y * w[24 + o] + z * w[48 + o];
        lift[g * 24 + o] = eluf(v);
    }
}

// -------------------- kernel 3: KNN (top-8 smallest distances) --------------------
__global__ __launch_bounds__(256) void knn_kernel(const float* __restrict__ pts,
                                                  int* __restrict__ idxOut) {
    __shared__ float sq[PP * 3];
    __shared__ float sr[PP];
    int n = blockIdx.x >> 5;
    int pblk = (blockIdx.x & 31) * 64;
    const float* bp = pts + (size_t)n * PP * 3;
    for (int t = threadIdx.x; t < PP; t += 256) {
        float x = bp[t * 3], y = bp[t * 3 + 1], z = bp[t * 3 + 2];
        sq[t * 3] = x; sq[t * 3 + 1] = y; sq[t * 3 + 2] = z;
        sr[t] = x * x + y * y + z * z;
    }
    __syncthreads();
    int w = threadIdx.x >> 5, lane = threadIdx.x & 31;
    for (int rep = 0; rep < 8; rep++) {
        int p = pblk + w * 8 + rep;
        float px = sq[p * 3], py = sq[p * 3 + 1], pz = sq[p * 3 + 2];
        float d[8]; int qi[8];
        #pragma unroll
        for (int j = 0; j < 8; j++) { d[j] = 3.4e38f; qi[j] = 0x7fffffff; }
        for (int q = lane; q < PP; q += 32) {
            float dist = sr[q] - 2.f * (px * sq[q * 3] + py * sq[q * 3 + 1] + pz * sq[q * 3 + 2]);
            if (dist < d[7] || (dist == d[7] && q < qi[7])) {
                d[7] = dist; qi[7] = q;
                #pragma unroll
                for (int j = 6; j >= 0; j--) {
                    bool sw = d[j + 1] < d[j] || (d[j + 1] == d[j] && qi[j + 1] < qi[j]);
                    if (sw) { float td = d[j]; d[j] = d[j + 1]; d[j + 1] = td;
                              int   ti = qi[j]; qi[j] = qi[j + 1]; qi[j + 1] = ti; }
                }
            }
        }
        int myout = 0;
        #pragma unroll
        for (int r = 0; r < 8; r++) {
            float md = d[0]; int mq = qi[0];
            #pragma unroll
            for (int off = 16; off >= 1; off >>= 1) {
                float od = __shfl_xor(md, off, 32);
                int   oq = __shfl_xor(mq, off, 32);
                if (od < md || (od == md && oq < mq)) { md = od; mq = oq; }
            }
            if (d[0] == md && qi[0] == mq) {           // pop my head
                #pragma unroll
                for (int j = 0; j < 7; j++) { d[j] = d[j + 1]; qi[j] = qi[j + 1]; }
                d[7] = 3.4e38f; qi[7] = 0x7fffffff;
            }
            if (lane == r) myout = mq;
        }
        if (lane < 8) idxOut[((size_t)(n * PP + p)) * 8 + lane] = myout;
    }
}

// -------------------- kernel 4: fused X-conv (one wave = 16 points) --------------------
__global__ __launch_bounds__(32) void xconv_kernel(
    const float* __restrict__ pts,
    const float* __restrict__ a1w1, const float* __restrict__ a1g1, const float* __restrict__ a1b1,
    const float* __restrict__ a1g2, const float* __restrict__ a1b2,
    const float* __restrict__ bxg0, const float* __restrict__ bxb0,
    const float* __restrict__ dw1,  const float* __restrict__ bxg1, const float* __restrict__ bxb1,
    const float* __restrict__ dw2,  const float* __restrict__ bxg2, const float* __restrict__ bxb2,
    const float* __restrict__ cdw,  const float* __restrict__ cg,   const float* __restrict__ cb,
    const _Float16* __restrict__ swz, const int* __restrict__ idx,
    const float* __restrict__ lift, float* __restrict__ out) {

    __shared__ __align__(16) _Float16 sFtsIn[128 * 48]; // rows=(pt,k): [0..23]=h, [24..47]=nn_fts_prev
    __shared__ __align__(16) _Float16 scr[9216];        // h1 / bxA / X / d / ftsX / dC staging
    __shared__ __align__(16) float    sLocal[128 * 3];  // local coords
    __shared__ __align__(16) float    sX[16 * 64];      // final X (f32)
    __shared__ __align__(16) _Float16 sDump[32];        // sink for masked-out columns

    int lane = threadIdx.x & 31;
    int nl = lane & 15, h = lane >> 4;
    int gp = blockIdx.x * 16;                           // global point base
    int n = gp >> 11, p0 = gp & 2047;
    const float* bpts = pts + (size_t)n * PP * 3;

    // ---- phase 0: gather local coords + neighbor lifted features ----
    for (int i = 0; i < 4; i++) {
        int s = lane + i * 32, pt = s >> 3;
        int q = idx[((size_t)(gp + pt)) * 8 + (s & 7)];
        float cx = bpts[(p0 + pt) * 3], cy = bpts[(p0 + pt) * 3 + 1], cz = bpts[(p0 + pt) * 3 + 2];
        sLocal[s * 3 + 0] = bpts[q * 3 + 0] - cx;
        sLocal[s * 3 + 1] = bpts[q * 3 + 1] - cy;
        sLocal[s * 3 + 2] = bpts[q * 3 + 2] - cz;
        const float* lf = lift + ((size_t)n * PP + q) * 24;
        #pragma unroll
        for (int j = 0; j < 24; j++) sFtsIn[s * 48 + 24 + j] = (_Float16)lf[j];
    }
    __builtin_amdgcn_wave_barrier();

    // ---- model_a1 layer 1 (K=3, VALU) -> h1 at scr[0..4095], stride 32 ----
    for (int i = 0; i < 4; i++) {
        int s = lane + i * 32;
        float x = sLocal[s * 3], y = sLocal[s * 3 + 1], z = sLocal[s * 3 + 2];
        #pragma unroll
        for (int o = 0; o < 24; o++) {
            float v = x * a1w1[o] + y * a1w1[24 + o] + z * a1w1[48 + o];
            scr[s * 32 + o] = (_Float16)(eluf(v) * (a1g1[o] * BNRS) + a1b1[o]);
        }
        #pragma unroll
        for (int o = 24; o < 32; o++) scr[s * 32 + o] = (_Float16)0.f;
    }
    __builtin_amdgcn_wave_barrier();

    // ---- model_a1 layer 2: [128,24]@[24,24] via WMMA -> sFtsIn[:,0..23] ----
    {
        v16h b0 = load_b(swz, 0, lane), b1 = load_b(swz, 1, lane);
        float sc0 = a1g2[nl] * BNRS, bb0 = a1b2[nl];
        int n1 = 16 + nl;
        bool live1 = (n1 < 24);
        float sc1 = live1 ? a1g2[n1] * BNRS : 0.f;
        float bb1 = live1 ? a1b2[n1] : 0.f;
        for (int mt = 0; mt < 8; mt++) {
            v16h a = load_a(scr, 32, mt * 16 + nl, 0, lane);
            v8f c0 = {}, c1 = {};
            c0 = wmma_f16(a, b0, c0);
            c1 = wmma_f16(a, b1, c1);
            #pragma unroll
            for (int g = 0; g < 8; g++) {
                int M = mt * 16 + h * 8 + g;
                sFtsIn[M * 48 + nl] = (_Float16)(eluf(c0[g]) * sc0 + bb0);
                // branch-free: dead columns (24..31) sink into sDump
                _Float16* dst = live1 ? &sFtsIn[M * 48 + n1] : &sDump[lane];
                *dst = (_Float16)(eluf(c1[g]) * sc1 + bb1);
            }
        }
    }
    __builtin_amdgcn_wave_barrier();

    // ---- bx: X0 = BN(elu([16,24]@[24,64])) via WMMA ----
    {
        _Float16* A = scr + 4096;                       // [16][32]
        #pragma unroll
        for (int j = 0; j < 16; j++) A[lane * 16 + j] = (_Float16)0.f;
        __builtin_amdgcn_wave_barrier();
        for (int i = 0; i < 4; i++) {
            int s = lane + i * 32, pt = s >> 3, k = s & 7;
            A[pt * 32 + k * 3 + 0] = (_Float16)sLocal[s * 3 + 0];
            A[pt * 32 + k * 3 + 1] = (_Float16)sLocal[s * 3 + 1];
            A[pt * 32 + k * 3 + 2] = (_Float16)sLocal[s * 3 + 2];
        }
        __builtin_amdgcn_wave_barrier();
        v16h a = load_a(scr + 4096, 32, nl, 0, lane);
        _Float16* X0 = scr + 4608;                      // [16][64]
        #pragma unroll
        for (int nt = 0; nt < 4; nt++) {
            v16h b = load_b(swz, 2 + nt, lane);
            v8f c = {};
            c = wmma_f16(a, b, c);
            int ncol = nt * 16 + nl;
            float sc = bxg0[ncol] * BNRS, bi = bxb0[ncol];
            #pragma unroll
            for (int g = 0; g < 8; g++)
                X0[(h * 8 + g) * 64 + ncol] = (_Float16)(eluf(c[g]) * sc + bi);
        }
    }
    __builtin_amdgcn_wave_barrier();

    // ---- sep1: depthwise (VALU) + pointwise (WMMA), elu+BN ----
    {
        const _Float16* Xin = scr + 4608;
        _Float16* D = scr;                              // [16][64], h1 is dead
        int pt = lane >> 1, ch = (lane & 1) * 4;
        for (int c = ch; c < ch + 4; c++) {
            float xv[8];
            #pragma unroll
            for (int k = 0; k < 8; k++) xv[k] = (float)Xin[pt * 64 + k * 8 + c];
            #pragma unroll
            for (int dd = 0; dd < 8; dd++) {
                float acc = 0.f;
                #pragma unroll
                for (int k = 0; k < 8; k++) acc += xv[k] * dw1[k * 64 + c * 8 + dd];
                D[pt * 64 + c * 8 + dd] = (_Float16)acc;
            }
        }
        __builtin_amdgcn_wave_barrier();
        v8f acc[4] = {};
        #pragma unroll
        for (int kt = 0; kt < 2; kt++) {
            v16h a = load_a(scr, 64, nl, kt * 32, lane);
            #pragma unroll
            for (int nt = 0; nt < 4; nt++) {
                v16h b = load_b(swz, 6 + kt * 4 + nt, lane);
                acc[nt] = wmma_f16(a, b, acc[nt]);
            }
        }
        __builtin_amdgcn_wave_barrier();
        _Float16* X1 = scr + 4608;
        #pragma unroll
        for (int nt = 0; nt < 4; nt++) {
            int ncol = nt * 16 + nl;
            float sc = bxg1[ncol] * BNRS, bi = bxb1[ncol];
            #pragma unroll
            for (int g = 0; g < 8; g++)
                X1[(h * 8 + g) * 64 + ncol] = (_Float16)(eluf(acc[nt][g]) * sc + bi);
        }
    }
    __builtin_amdgcn_wave_barrier();

    // ---- sep2: depthwise + pointwise, BN only (no activation) -> sX (f32) ----
    {
        const _Float16* Xin = scr + 4608;
        _Float16* D = scr;
        int pt = lane >> 1, ch = (lane & 1) * 4;
        for (int c = ch; c < ch + 4; c++) {
            float xv[8];
            #pragma unroll
            for (int k = 0; k < 8; k++) xv[k] = (float)Xin[pt * 64 + k * 8 + c];
            #pragma unroll
            for (int dd = 0; dd < 8; dd++) {
                float acc = 0.f;
                #pragma unroll
                for (int k = 0; k < 8; k++) acc += xv[k] * dw2[k * 64 + c * 8 + dd];
                D[pt * 64 + c * 8 + dd] = (_Float16)acc;
            }
        }
        __builtin_amdgcn_wave_barrier();
        v8f acc[4] = {};
        #pragma unroll
        for (int kt = 0; kt < 2; kt++) {
            v16h a = load_a(scr, 64, nl, kt * 32, lane);
            #pragma unroll
            for (int nt = 0; nt < 4; nt++) {
                v16h b = load_b(swz, 14 + kt * 4 + nt, lane);
                acc[nt] = wmma_f16(a, b, acc[nt]);
            }
        }
        #pragma unroll
        for (int nt = 0; nt < 4; nt++) {
            int ncol = nt * 16 + nl;
            float sc = bxg2[ncol] * BNRS, bi = bxb2[ncol];
            #pragma unroll
            for (int g = 0; g < 8; g++)
                sX[(h * 8 + g) * 64 + ncol] = acc[nt][g] * sc + bi;
        }
    }
    __builtin_amdgcn_wave_barrier();

    // ---- fts_X[pt][k][c] = sum_j X[pt][k][j] * ftsIn[pt][j][c]  (VALU) -> scr[0..6143] ----
    for (int i = 0; i < 4; i++) {
        int s = lane + i * 32, pt = s >> 3, k = s & 7;
        float accs[48];
        #pragma unroll
        for (int c = 0; c < 48; c++) accs[c] = 0.f;
        #pragma unroll
        for (int j = 0; j < 8; j++) {
            float xv = sX[pt * 64 + k * 8 + j];
            const _Float16* fr = sFtsIn + (pt * 8 + j) * 48;
            #pragma unroll
            for (int c = 0; c < 48; c++) accs[c] += xv * (float)fr[c];
        }
        _Float16* fx = scr + s * 48;
        #pragma unroll
        for (int c = 0; c < 48; c++) fx[c] = (_Float16)accs[c];
    }
    __builtin_amdgcn_wave_barrier();

    // ---- model_c depthwise (VALU) -> dC at scr[6144..9215], [16][192] ----
    {
        int pt = lane >> 1, cbase = (lane & 1) * 24;
        _Float16* DC = scr + 6144;
        for (int c = cbase; c < cbase + 24; c++) {
            float fv[8];
            #pragma unroll
            for (int k = 0; k < 8; k++) fv[k] = (float)scr[(pt * 8 + k) * 48 + c];
            #pragma unroll
            for (int dd = 0; dd < 4; dd++) {
                float acc = 0.f;
                #pragma unroll
                for (int k = 0; k < 8; k++) acc += fv[k] * cdw[k * 192 + c * 4 + dd];
                DC[pt * 192 + c * 4 + dd] = (_Float16)acc;
            }
        }
    }
    __builtin_amdgcn_wave_barrier();

    // ---- model_c pointwise: [16,192]@[192,48] via WMMA, elu+BN -> out ----
    {
        v8f acc[3] = {};
        #pragma unroll
        for (int kt = 0; kt < 6; kt++) {
            v16h a = load_a(scr + 6144, 192, nl, kt * 32, lane);
            #pragma unroll
            for (int nt = 0; nt < 3; nt++) {
                v16h b = load_b(swz, 22 + kt * 3 + nt, lane);
                acc[nt] = wmma_f16(a, b, acc[nt]);
            }
        }
        #pragma unroll
        for (int nt = 0; nt < 3; nt++) {
            int ncol = nt * 16 + nl;
            float sc = cg[ncol] * BNRS, bi = cb[ncol];
            #pragma unroll
            for (int g = 0; g < 8; g++) {
                int pt = h * 8 + g;
                out[((size_t)(n * PP) + p0 + pt) * 48 + ncol] = eluf(acc[nt][g]) * sc + bi;
            }
        }
    }
}

extern "C" void kernel_launch(void* const* d_in, const int* in_sizes, int n_in,
                              void* d_out, int out_size, void* d_ws, size_t ws_size,
                              hipStream_t stream) {
    const float* pts     = (const float*)d_in[0];
    const float* fts     = (const float*)d_in[1];
    const float* dense0w = (const float*)d_in[2];
    const float* a1_w1   = (const float*)d_in[3];
    const float* a1_g1   = (const float*)d_in[4];
    const float* a1_b1   = (const float*)d_in[5];
    const float* a1_w2   = (const float*)d_in[6];
    const float* a1_g2   = (const float*)d_in[7];
    const float* a1_b2   = (const float*)d_in[8];
    const float* bx_w    = (const float*)d_in[9];
    const float* bx_g0   = (const float*)d_in[10];
    const float* bx_b0   = (const float*)d_in[11];
    const float* bd1_dw  = (const float*)d_in[12];
    const float* bd1_pw  = (const float*)d_in[13];
    const float* bx_g1   = (const float*)d_in[14];
    const float* bx_b1   = (const float*)d_in[15];
    const float* bd2_dw  = (const float*)d_in[16];
    const float* bd2_pw  = (const float*)d_in[17];
    const float* bx_g2   = (const float*)d_in[18];
    const float* bx_b2   = (const float*)d_in[19];
    const float* c_dw    = (const float*)d_in[20];
    const float* c_pw    = (const float*)d_in[21];
    const float* c_g     = (const float*)d_in[22];
    const float* c_b     = (const float*)d_in[23];

    char* ws = (char*)d_ws;
    _Float16* swz = (_Float16*)ws;
    int*      idx = (int*)(ws + WS_IDX_OFF);
    float*    lft = (float*)(ws + WS_LIFT_OFF);
    float*    out = (float*)d_out;

    prep_weights<<<80, 256, 0, stream>>>(a1_w2, bx_w, bd1_pw, bd2_pw, c_pw, swz);
    lift_kernel<<<NPTS / 256, 256, 0, stream>>>(fts, dense0w, lft);
    knn_kernel<<<1024, 256, 0, stream>>>(pts, idx);
    xconv_kernel<<<NPTS / 16, 32, 0, stream>>>(pts, a1_w1, a1_g1, a1_b1, a1_g2, a1_b2,
                                               bx_g0, bx_b0, bd1_dw, bx_g1, bx_b1,
                                               bd2_dw, bx_g2, bx_b2, c_dw, c_g, c_b,
                                               swz, idx, lft, out);
}
